// LSTM_72928544686170
// MI455X (gfx1250) — matrix operand order repaired and checked by reading, not compiled, not run on hardware
//
#include <hip/hip_runtime.h>
#include <hip/hip_bf16.h>

typedef __bf16 bf16;
typedef __attribute__((ext_vector_type(16))) __bf16 v16bf;
typedef __attribute__((ext_vector_type(8)))  float  v8f;

#define BTOT   16384
#define TSEQ   60
#define TCH    30
#define DEMB   50
#define HDIM   64
#define GDIM   256      // 4*H
#define LDK    136      // padded LDS row stride (bf16 elems), even, 68 dwords -> low bank conflict

#define LOG2E  1.4426950408889634f

union AFrag { v16bf v; unsigned u[8]; };

// Branch-free gate math on native v_exp_f32 / v_rcp_f32 (exp2 clamps to 0/+inf in HW,
// so both functions saturate correctly without libm's divergent fixup paths).
__device__ __forceinline__ float fast_sigmoid(float x) {
    return __builtin_amdgcn_rcpf(1.0f + __builtin_amdgcn_exp2f(-x * LOG2E));
}
__device__ __forceinline__ float fast_tanh(float x) {
    return 1.0f - 2.0f * __builtin_amdgcn_rcpf(1.0f + __builtin_amdgcn_exp2f(x * (2.0f * LOG2E)));
}

__global__ __launch_bounds__(128) void lstm_chunk_kernel(
    const float* __restrict__ x, const float* __restrict__ Wih,
    const float* __restrict__ Whh, const float* __restrict__ bih,
    const float* __restrict__ bhh, float* __restrict__ hbuf)
{
    __shared__ bf16 xh[16][LDK];   // A staging: [m][k], k in [0,128): 0..49 x, 50..63 zero, 64..127 h

    const int tid    = threadIdx.x;
    const int lane   = tid & 31;
    const int wave   = tid >> 5;          // 0..3
    const int laneHi = lane >> 4;         // 0/1
    const int ln     = lane & 15;
    const int b0     = blockIdx.x * 16;   // batch tile base
    const int chunk  = blockIdx.y;        // 0/1
    const int t0     = chunk * TCH;

    // ---- Preload B fragments (shared across all 30 timesteps) into registers ----
    // B element e of fragment (tile,s): n = tile*64 + wave*16 + ln ; k = s*32 + laneHi*16 + e
    v16bf Bf[4][4];
    float biasv[4];
    #pragma unroll
    for (int tile = 0; tile < 4; ++tile) {
        const int n = tile * 64 + wave * 16 + ln;
        biasv[tile] = bih[n] + bhh[n];
        #pragma unroll
        for (int s = 0; s < 4; ++s) {
            #pragma unroll
            for (int e = 0; e < 16; ++e) {
                const int k = s * 32 + laneHi * 16 + e;
                float w;
                if (k < HDIM)  w = (k < DEMB) ? Wih[n * DEMB + k] : 0.0f;
                else           w = Whh[n * HDIM + (k - HDIM)];
                Bf[tile][s][e] = (bf16)w;
            }
        }
    }

    // ---- Init LDS: zero everything (x pad + h(t=0)=0), stage x(t=0) ----
    for (int idx = tid; idx < 16 * LDK; idx += 128)
        xh[idx / LDK][idx % LDK] = (bf16)0.0f;
    __syncthreads();
    for (int idx = tid; idx < 16 * DEMB; idx += 128) {
        const int m = idx / DEMB, d = idx % DEMB;
        xh[m][d] = (bf16)x[((size_t)(b0 + m) * TSEQ + t0) * DEMB + d];
    }
    __syncthreads();

    float cst[8];
    #pragma unroll
    for (int r = 0; r < 8; ++r) cst[r] = 0.0f;

    const int nh = wave * 16 + ln;        // this lane's hidden column [0,64)

    for (int t = 0; t < TCH; ++t) {
        // ---- Gather A fragments from LDS (pairs as aligned b32) ----
        // element pair v of fragment s: K = s*32 + laneHi*16 + 2v
        AFrag Af[4];
        {
            const unsigned* rowp = (const unsigned*)&xh[ln][0];   // row base is 4B aligned (272B stride)
            #pragma unroll
            for (int s = 0; s < 4; ++s) {
                const int kb2 = (s * 32 + laneHi * 16) >> 1;      // dword index base
                #pragma unroll
                for (int v = 0; v < 8; ++v)
                    Af[s].u[v] = rowp[kb2 + v];
            }
        }
        __syncthreads();   // all xh reads complete before anyone rewrites it

        // ---- 4 gate tiles x 4 k-steps of v_wmma_f32_16x16x32_bf16 ----
        v8f acc[4];
        #pragma unroll
        for (int tile = 0; tile < 4; ++tile) {
            #pragma unroll
            for (int r = 0; r < 8; ++r) acc[tile][r] = biasv[tile];
            #pragma unroll
            for (int s = 0; s < 4; ++s)
                acc[tile] = __builtin_amdgcn_wmma_f32_16x16x32_bf16(
                    false, Af[s].v, false, Bf[tile][s], (short)0, acc[tile], false, false);
        }

        // ---- Elementwise gates, state update, write h back to LDS (branch-free) ----
        #pragma unroll
        for (int r = 0; r < 8; ++r) {
            const float iv = fast_sigmoid(acc[0][r]);
            const float fv = fast_sigmoid(acc[1][r]);
            const float gv = fast_tanh(acc[2][r]);
            const float ov = fast_sigmoid(acc[3][r]);
            const float cv = fv * cst[r] + iv * gv;
            cst[r] = cv;
            const float hv = ov * fast_tanh(cv);
            const int m = r + 8 * laneHi;
            xh[m][HDIM + nh] = (bf16)hv;
            if (t == TCH - 1)
                hbuf[(size_t)(b0 + m) * 128 + chunk * 64 + nh] = hv;
        }

        // ---- Stage x(t+1) ----
        if (t + 1 < TCH) {
            for (int idx = tid; idx < 16 * DEMB; idx += 128) {
                const int m = idx / DEMB, d = idx % DEMB;
                xh[m][d] = (bf16)x[((size_t)(b0 + m) * TSEQ + (t0 + t + 1)) * DEMB + d];
            }
        }
        __syncthreads();   // writes visible before next iteration's reads
    }
}

__global__ __launch_bounds__(256) void mlp_head_kernel(
    const float* __restrict__ hbuf,
    const float* __restrict__ W1, const float* __restrict__ b1,
    const float* __restrict__ W2, const float* __restrict__ b2,
    const float* __restrict__ W3, const float* __restrict__ b3,
    float* __restrict__ out)
{
    __shared__ float sW1[32][128];
    __shared__ float sW2[32][32];
    __shared__ float sW3[32], sb1[32], sb2[32];
    __shared__ float sb3;

    const int tid = threadIdx.x;
    for (int idx = tid; idx < 32 * 128; idx += 256) sW1[idx >> 7][idx & 127] = W1[idx];
    for (int idx = tid; idx < 32 * 32;  idx += 256) sW2[idx >> 5][idx & 31]  = W2[idx];
    if (tid < 32) { sW3[tid] = W3[tid]; sb1[tid] = b1[tid]; sb2[tid] = b2[tid]; }
    if (tid == 0) sb3 = b3[0];
    __syncthreads();

    const int b = blockIdx.x * 256 + tid;
    const float* z = hbuf + (size_t)b * 128;

    float a1[32];
    #pragma unroll
    for (int j = 0; j < 32; ++j) a1[j] = sb1[j];
    for (int k = 0; k < 128; ++k) {
        const float zk = z[k];
        #pragma unroll
        for (int j = 0; j < 32; ++j) a1[j] = fmaf(zk, sW1[j][k], a1[j]);
    }
    #pragma unroll
    for (int j = 0; j < 32; ++j) a1[j] = fmaxf(a1[j], 0.0f);

    float a2[32];
    #pragma unroll
    for (int j = 0; j < 32; ++j) a2[j] = sb2[j];
    #pragma unroll
    for (int k = 0; k < 32; ++k) {
        const float ak = a1[k];
        #pragma unroll
        for (int j = 0; j < 32; ++j) a2[j] = fmaf(ak, sW2[j][k], a2[j]);
    }

    float s = sb3;
    #pragma unroll
    for (int j = 0; j < 32; ++j) s = fmaf(fmaxf(a2[j], 0.0f), sW3[j], s);
    out[b] = fast_sigmoid(s);
}

extern "C" void kernel_launch(void* const* d_in, const int* in_sizes, int n_in,
                              void* d_out, int out_size, void* d_ws, size_t ws_size,
                              hipStream_t stream) {
    (void)in_sizes; (void)n_in; (void)out_size; (void)ws_size;
    const float* x    = (const float*)d_in[0];
    const float* Wih  = (const float*)d_in[1];
    const float* Whh  = (const float*)d_in[2];
    const float* bih  = (const float*)d_in[3];
    const float* bhh  = (const float*)d_in[4];
    const float* W1   = (const float*)d_in[5];
    const float* b1   = (const float*)d_in[6];
    const float* W2   = (const float*)d_in[7];
    const float* b2   = (const float*)d_in[8];
    const float* W3   = (const float*)d_in[9];
    const float* b3   = (const float*)d_in[10];
    float* hbuf = (float*)d_ws;             // [B][128] f32 = 8 MB
    float* outp = (float*)d_out;

    dim3 grid1(BTOT / 16, 2);
    lstm_chunk_kernel<<<grid1, 128, 0, stream>>>(x, Wih, Whh, bih, bhh, hbuf);
    mlp_head_kernel<<<BTOT / 256, 256, 0, stream>>>(hbuf, W1, b1, W2, b2, W3, b3, outp);
}
